// BLTLocalEncoder_68676527063271
// MI455X (gfx1250) — matrix operand order repaired and checked by reading, not compiled, not run on hardware
//
#include <hip/hip_runtime.h>
#include <hip/hip_bf16.h>

// ---------------------------------------------------------------------------
// BLT local encoder for MI455X (gfx1250), wave32 + WMMA bf16.
// All matmuls use v_wmma_f32_16x16x32_bf16; softmax/norm math in f32.
// GEMM: 32x64 register-blocked macro-tile per wave, double-buffered K-loop
// (loads for step k+32 in flight while WMMAs consume step k).
// ---------------------------------------------------------------------------

typedef __bf16 bf16;
typedef __attribute__((ext_vector_type(16))) __bf16 v16bf;
typedef __attribute__((ext_vector_type(8)))  __bf16 v8bf;
typedef __attribute__((ext_vector_type(8)))  float  v8f;

#define CB 4
#define CS 2048
#define CP 512
#define CD 512
#define CNH 8
#define CDH 64
#define CF 1536
#define CHASH_VOCAB 50002

static __device__ __forceinline__ v16bf cat8(v8bf lo, v8bf hi) {
  return __builtin_shufflevector(lo, hi, 0,1,2,3,4,5,6,7,8,9,10,11,12,13,14,15);
}

// ---------------------------------------------------------------------------
// 1. Hash embeddings: h[b,s,:] = tok_emb[tok] + sum_t hash_emb[t][hash_t]
// ---------------------------------------------------------------------------
__global__ __launch_bounds__(128) void hash_embed_kernel(
    const int* __restrict__ tokens, const float* __restrict__ tok_emb,
    const float* __restrict__ hash_emb, float* __restrict__ h) {
  int bs = blockIdx.x;
  int b = bs / CS, s = bs % CS;
  const int* trow = tokens + (size_t)b * CS;
  int tok = trow[s];
  const long long primes[2] = {1000000007LL, 5915587277LL};
  const int gs[2] = {3, 4};
  int hidx[4];
  int t = 0;
  #pragma unroll
  for (int pi = 0; pi < 2; ++pi) {
    int p = (int)(primes[pi] % (long long)CHASH_VOCAB);
    #pragma unroll
    for (int gi = 0; gi < 2; ++gi) {
      int g = gs[gi];
      int hsh = 0, pw = 1;
      for (int i = 0; i < g; ++i) {
        int pos = s - (g - 1) + i;
        int tv = (pos >= 0) ? trow[pos] : 0;
        hsh = (int)((hsh + (long long)tv * pw) % CHASH_VOCAB);
        pw  = (int)(((long long)pw * p) % CHASH_VOCAB);
      }
      hidx[t++] = hsh;
    }
  }
  float* out = h + ((size_t)b * CS + s) * CD;
  const float* te = tok_emb + (size_t)tok * CD;
  for (int d = threadIdx.x; d < CD; d += 128) {
    float acc = te[d];
    #pragma unroll
    for (int j = 0; j < 4; ++j)
      acc += hash_emb[((size_t)j * CHASH_VOCAB + hidx[j]) * CD + d];
    out[d] = acc;
  }
}

// ---------------------------------------------------------------------------
// 2. RMS norm (f32 rows) -> bf16
// ---------------------------------------------------------------------------
__global__ __launch_bounds__(256) void rms_cast_kernel(
    const float* __restrict__ x, const float* __restrict__ w,
    bf16* __restrict__ y, int D) {
  int row = blockIdx.x;
  const float* xr = x + (size_t)row * D;
  float ss = 0.f;
  for (int i = threadIdx.x; i < D; i += 256) { float v = xr[i]; ss += v * v; }
  #pragma unroll
  for (int off = 16; off > 0; off >>= 1) ss += __shfl_xor(ss, off, 32);
  __shared__ float sbuf[8];
  if ((threadIdx.x & 31) == 0) sbuf[threadIdx.x >> 5] = ss;
  __syncthreads();
  float tot = 0.f;
  #pragma unroll
  for (int i = 0; i < 8; ++i) tot += sbuf[i];
  float rinv = rsqrtf(tot / (float)D + 1e-5f);
  for (int i = threadIdx.x; i < D; i += 256)
    y[(size_t)row * D + i] = (bf16)(xr[i] * rinv * w[i]);
}

// ---------------------------------------------------------------------------
// 3. Weight transpose + cast: W (K x N) f32 -> Wt (N x K) bf16
// ---------------------------------------------------------------------------
__global__ __launch_bounds__(256) void transpose_cast_w_kernel(
    const float* __restrict__ W, bf16* __restrict__ Wt, int K, int N) {
  int i = blockIdx.x * 256 + threadIdx.x;
  if (i >= K * N) return;
  int n = i / K, k = i % K;
  Wt[(size_t)n * K + k] = (bf16)W[(size_t)k * N + n];
}

// ---------------------------------------------------------------------------
// 4. WMMA bf16 GEMM, register-blocked 32x64 per wave, double-buffered K-loop.
//    C[M,N] = A[M,K] * Bt[N,K]^T (+ optional residual Csrc).
//    Requires K % 64 == 0 (true for K in {512, 1536}).
// ---------------------------------------------------------------------------
#define GEMM_MT 2   // 16-row subtiles per wave
#define GEMM_NT 4   // 16-col subtiles per wave

#define LOAD_FRAGS(AF, BF, K0)                                        \
  do {                                                                \
    _Pragma("unroll")                                                 \
    for (int mi = 0; mi < GEMM_MT; ++mi) {                            \
      v8bf lo = *(const v8bf*)(arow[mi] + (K0));                      \
      v8bf hi = *(const v8bf*)(arow[mi] + (K0) + 8);                  \
      AF[mi] = cat8(lo, hi);                                          \
    }                                                                 \
    _Pragma("unroll")                                                 \
    for (int ni = 0; ni < GEMM_NT; ++ni)                              \
      BF[ni] = *(const v16bf*)(brow[ni] + (K0));                      \
  } while (0)

#define DO_WMMA(AF, BF)                                               \
  do {                                                                \
    _Pragma("unroll")                                                 \
    for (int mi = 0; mi < GEMM_MT; ++mi)                              \
      _Pragma("unroll")                                               \
      for (int ni = 0; ni < GEMM_NT; ++ni)                            \
        acc[mi][ni] = __builtin_amdgcn_wmma_f32_16x16x32_bf16(        \
            false, AF[mi], false, BF[ni], (short)0, acc[mi][ni],      \
            false, false);                                            \
  } while (0)

__global__ __launch_bounds__(256) void gemm_wmma_bf16_kernel(
    const bf16* __restrict__ A, const bf16* __restrict__ Bt,
    const float* __restrict__ Csrc, float* __restrict__ C,
    int M, int N, int K) {
  int wave = threadIdx.x >> 5;
  int lane = threadIdx.x & 31;
  int half = lane >> 4;
  int l16  = lane & 15;
  int tilesM = M / (16 * GEMM_MT), tilesN = N / (16 * GEMM_NT);
  int tid = blockIdx.x * 8 + wave;
  if (tid >= tilesM * tilesN) return;
  int tM = tid / tilesN, tN = tid % tilesN;

  v8f acc[GEMM_MT][GEMM_NT];
  #pragma unroll
  for (int mi = 0; mi < GEMM_MT; ++mi)
    #pragma unroll
    for (int ni = 0; ni < GEMM_NT; ++ni) acc[mi][ni] = v8f{};

  const bf16* arow[GEMM_MT];
  const bf16* brow[GEMM_NT];
  #pragma unroll
  for (int mi = 0; mi < GEMM_MT; ++mi)
    arow[mi] = A + (size_t)(tM * 16 * GEMM_MT + mi * 16 + l16) * K + half * 16;
  #pragma unroll
  for (int ni = 0; ni < GEMM_NT; ++ni)
    brow[ni] = Bt + (size_t)(tN * 16 * GEMM_NT + ni * 16 + l16) * K + half * 16;

  v16bf af0[GEMM_MT], bf0[GEMM_NT];
  v16bf af1[GEMM_MT], bf1[GEMM_NT];

  LOAD_FRAGS(af0, bf0, 0);
  for (int k0 = 0; k0 < K; k0 += 64) {
    // loads for k0+32 fly while WMMAs consume k0
    LOAD_FRAGS(af1, bf1, k0 + 32);
    #pragma unroll
    for (int mi = 0; mi < GEMM_MT; ++mi)
      __builtin_prefetch(arow[mi] + k0 + 128, 0, 1);
    DO_WMMA(af0, bf0);
    if (k0 + 64 < K) LOAD_FRAGS(af0, bf0, k0 + 64);
    DO_WMMA(af1, bf1);
  }

  #pragma unroll
  for (int mi = 0; mi < GEMM_MT; ++mi) {
    size_t rbase = (size_t)(tM * 16 * GEMM_MT + mi * 16 + half * 8) * N
                 + (size_t)tN * 16 * GEMM_NT + l16;
    #pragma unroll
    for (int ni = 0; ni < GEMM_NT; ++ni) {
      #pragma unroll
      for (int v = 0; v < 8; ++v) {
        size_t idx = rbase + (size_t)v * N + ni * 16;
        float r = acc[mi][ni][v];
        if (Csrc) r += Csrc[idx];
        C[idx] = r;
      }
    }
  }
}

// ---------------------------------------------------------------------------
// 5. RoPE (f32 in, bf16 out), layout (B,S,NH,DH); i enumerates (b,s,h,pair)
// ---------------------------------------------------------------------------
__global__ __launch_bounds__(256) void rope_cast_kernel(
    const float* __restrict__ x, bf16* __restrict__ y, int S, long long n) {
  long long i = (long long)blockIdx.x * 256 + threadIdx.x;
  if (i >= n) return;
  int pr = (int)(i & 31);
  int s  = (int)((i >> 8) % S);
  float freq = __powf(10000.f, -(float)pr / 32.f);
  float ang = (float)s * freq;
  float c = __cosf(ang), sn = __sinf(ang);
  float x1 = x[2 * i], x2 = x[2 * i + 1];
  y[2 * i]     = (bf16)(x1 * c - x2 * sn);
  y[2 * i + 1] = (bf16)(x1 * sn + x2 * c);
}

// ---------------------------------------------------------------------------
// 6. V transpose: (B,S,NH,DH) f32 -> (B,NH,DH,S) bf16
// ---------------------------------------------------------------------------
__global__ __launch_bounds__(256) void transpose_v_kernel(
    const float* __restrict__ x, bf16* __restrict__ y, int S, long long n) {
  long long i = (long long)blockIdx.x * 256 + threadIdx.x;
  if (i >= n) return;
  int s = (int)(i % S);
  long long r = i / S;
  int d  = (int)(r & 63);
  int hh = (int)((r >> 6) & 7);
  int b  = (int)(r >> 9);
  y[i] = (bf16)x[(((long long)b * S + s) * CNH + hh) * CDH + d];
}

__global__ __launch_bounds__(256) void cast_bf_kernel(
    const float* __restrict__ x, bf16* __restrict__ y, long long n) {
  long long i = (long long)blockIdx.x * 256 + threadIdx.x;
  if (i >= n) return;
  y[i] = (bf16)x[i];
}

__global__ __launch_bounds__(256) void silu_mul_kernel(
    const float* __restrict__ a, const float* __restrict__ b,
    bf16* __restrict__ y, long long n) {
  long long i = (long long)blockIdx.x * 256 + threadIdx.x;
  if (i >= n) return;
  float av = a[i];
  float s = av / (1.f + __expf(-av));
  y[i] = (bf16)(s * b[i]);
}

// ---------------------------------------------------------------------------
// 7. Flash attention with WMMA. One wave per (b, head, 16-query tile).
//    Q,K: (B,Sq/Skv,NH,DH) bf16; VT: (B,NH,DH,Skv) bf16; O: (B,Sq,NH*DH) f32
// ---------------------------------------------------------------------------
__global__ __launch_bounds__(32) void attn_wmma_kernel(
    const bf16* __restrict__ Q, const bf16* __restrict__ Kk,
    const bf16* __restrict__ VT, float* __restrict__ O,
    int Sq, int Skv, int causal) {
  __shared__ bf16 pbuf[16 * 32];
  int lane = threadIdx.x & 31;
  int half = lane >> 4;
  int l16  = lane & 15;
  int qt = blockIdx.x, hh = blockIdx.y, b = blockIdx.z;
  int qbase = qt * 16;
  const float scale = 0.125f;  // 1/sqrt(64)

  // Q fragments for d0 = 0 and 32 (K-dim of the score WMMA)
  v16bf qf[2];
  const bf16* qrow = Q + (((size_t)b * Sq + qbase + l16) * CNH + hh) * CDH + half * 16;
  #pragma unroll
  for (int i = 0; i < 2; ++i) {
    v8bf lo = *(const v8bf*)(qrow + i * 32);
    v8bf hi = *(const v8bf*)(qrow + i * 32 + 8);
    qf[i] = cat8(lo, hi);
  }

  v8f oacc[4] = {v8f{}, v8f{}, v8f{}, v8f{}};
  float mrow[8], lrow[8];
  #pragma unroll
  for (int v = 0; v < 8; ++v) { mrow[v] = -1e30f; lrow[v] = 0.f; }

  int kend = causal ? (qbase + 16) : Skv;
  const bf16* vtb = VT + ((size_t)(b * CNH + hh) * CDH) * Skv;

  for (int kb = 0; kb < kend; kb += 32) {
    // ---- scores: two 16-key blocks, each = 2 WMMA over DH split
    v8f s0 = {}, s1 = {};
    {
      const bf16* krow = Kk + (((size_t)b * Skv + kb + l16) * CNH + hh) * CDH + half * 16;
      v16bf bd0 = *(const v16bf*)(krow);
      v16bf bd1 = *(const v16bf*)(krow + 32);
      s0 = __builtin_amdgcn_wmma_f32_16x16x32_bf16(false, qf[0], false, bd0, (short)0, s0, false, false);
      s0 = __builtin_amdgcn_wmma_f32_16x16x32_bf16(false, qf[1], false, bd1, (short)0, s0, false, false);
    }
    {
      const bf16* krow = Kk + (((size_t)b * Skv + kb + 16 + l16) * CNH + hh) * CDH + half * 16;
      v16bf bd0 = *(const v16bf*)(krow);
      v16bf bd1 = *(const v16bf*)(krow + 32);
      s1 = __builtin_amdgcn_wmma_f32_16x16x32_bf16(false, qf[0], false, bd0, (short)0, s1, false, false);
      s1 = __builtin_amdgcn_wmma_f32_16x16x32_bf16(false, qf[1], false, bd1, (short)0, s1, false, false);
    }
    // ---- online softmax (row = v + 8*half, cols striped over 16 lanes)
    #pragma unroll
    for (int v = 0; v < 8; ++v) {
      int qg = qbase + v + 8 * half;
      float a0 = s0[v] * scale, a1 = s1[v] * scale;
      if (causal) {
        if (kb + l16 > qg)      a0 = -1e30f;
        if (kb + 16 + l16 > qg) a1 = -1e30f;
      }
      float mloc = fmaxf(a0, a1);
      #pragma unroll
      for (int off = 1; off < 16; off <<= 1) mloc = fmaxf(mloc, __shfl_xor(mloc, off, 16));
      float mnew = fmaxf(mrow[v], mloc);
      float corr = __expf(mrow[v] - mnew);
      mrow[v] = mnew;
      float p0 = __expf(a0 - mnew), p1 = __expf(a1 - mnew);
      float ls = p0 + p1;
      #pragma unroll
      for (int off = 1; off < 16; off <<= 1) ls += __shfl_xor(ls, off, 16);
      lrow[v] = lrow[v] * corr + ls;
      #pragma unroll
      for (int nb = 0; nb < 4; ++nb) oacc[nb][v] *= corr;
      pbuf[(v + 8 * half) * 32 + l16]      = (bf16)p0;
      pbuf[(v + 8 * half) * 32 + 16 + l16] = (bf16)p1;
    }
    __syncthreads();
    // ---- P as A-fragment from LDS
    const bf16* prow = pbuf + l16 * 32 + half * 16;
    v8bf plo = *(const v8bf*)(prow);
    v8bf phi = *(const v8bf*)(prow + 8);
    v16bf pf = cat8(plo, phi);
    // ---- P @ V : K=32 keys, 4 column blocks of 16
    #pragma unroll
    for (int nb = 0; nb < 4; ++nb) {
      v16bf vfm = *(const v16bf*)(vtb + (size_t)(nb * 16 + l16) * Skv + kb + half * 16);
      oacc[nb] = __builtin_amdgcn_wmma_f32_16x16x32_bf16(
          false, pf, false, vfm, (short)0, oacc[nb], false, false);
    }
    __syncthreads();
  }

  // ---- epilogue: divide by row sum, store (B,Sq,NH,DH)
  #pragma unroll
  for (int v = 0; v < 8; ++v) {
    float inv = 1.f / lrow[v];
    size_t rowbase = (((size_t)b * Sq + qbase + v + 8 * half) * CNH + hh) * CDH + l16;
    #pragma unroll
    for (int nb = 0; nb < 4; ++nb)
      O[rowbase + nb * 16] = oacc[nb][v] * inv;
  }
}

// ---------------------------------------------------------------------------
// 8. Patch pooling: cumsum, segment-max scatter, finalize
// ---------------------------------------------------------------------------
__global__ __launch_bounds__(512) void cumsum_kernel(
    const int* __restrict__ pl, int* __restrict__ cum) {
  __shared__ int buf[CP];
  int b = blockIdx.x, t = threadIdx.x;
  buf[t] = pl[(size_t)b * CP + t];
  __syncthreads();
  for (int off = 1; off < CP; off <<= 1) {
    int v = (t >= off) ? buf[t - off] : 0;
    __syncthreads();
    buf[t] += v;
    __syncthreads();
  }
  cum[(size_t)b * CP + t] = buf[t];
}

__global__ __launch_bounds__(256) void fill_neg_inf_kernel(float* __restrict__ p, long long n) {
  long long i = (long long)blockIdx.x * 256 + threadIdx.x;
  if (i < n) p[i] = __int_as_float(0xFF800000u);
}

static __device__ __forceinline__ void atomicMaxF(float* addr, float val) {
  if (val >= 0.f) atomicMax((int*)addr, __float_as_int(val));
  else            atomicMin((unsigned int*)addr, (unsigned int)__float_as_int(val));
}

__global__ __launch_bounds__(128) void scatter_max_kernel(
    const float* __restrict__ h, const int* __restrict__ cum,
    float* __restrict__ pooled) {
  int bs = blockIdx.x;
  int b = bs / CS, s = bs % CS;
  const int* c = cum + (size_t)b * CP;
  int lo = 0, hi = CP;
  while (lo < hi) { int mid = (lo + hi) >> 1; if (c[mid] > s) hi = mid; else lo = mid + 1; }
  int pid = min(lo, CP - 1);
  const float* src = h + ((size_t)b * CS + s) * CD;
  float* dst = pooled + ((size_t)b * CP + pid) * CD;
  for (int d = threadIdx.x; d < CD; d += 128) atomicMaxF(dst + d, src[d]);
}

__global__ __launch_bounds__(256) void fix_pooled_kernel(float* __restrict__ p, long long n) {
  long long i = (long long)blockIdx.x * 256 + threadIdx.x;
  if (i >= n) return;
  float v = p[i];
  if (!(v > -1e37f)) p[i] = 0.f;
}

// ---------------------------------------------------------------------------
// Host-side orchestration
// ---------------------------------------------------------------------------
static inline int cdiv(long long a, long long b) { return (int)((a + b - 1) / b); }

extern "C" void kernel_launch(void* const* d_in, const int* in_sizes, int n_in,
                              void* d_out, int out_size, void* d_ws, size_t ws_size,
                              hipStream_t stream) {
  (void)in_sizes; (void)n_in; (void)out_size; (void)ws_size;
  const int*   tokens    = (const int*)d_in[0];
  const int*   patch_len = (const int*)d_in[1];
  const float* tok_emb   = (const float*)d_in[2];
  const float* hash_emb  = (const float*)d_in[3];
  const float* wq        = (const float*)d_in[4];
  const float* wk        = (const float*)d_in[5];
  const float* wv        = (const float*)d_in[6];
  const float* wo        = (const float*)d_in[7];
  const float* attn_nw   = (const float*)d_in[8];
  const float* ffn_nw    = (const float*)d_in[9];
  const float* w1        = (const float*)d_in[10];
  const float* w3        = (const float*)d_in[11];
  const float* w2        = (const float*)d_in[12];
  const float* cq_nw     = (const float*)d_in[13];
  const float* ckv_nw    = (const float*)d_in[14];
  const float* cwq       = (const float*)d_in[15];
  const float* cwk       = (const float*)d_in[16];
  const float* cwv       = (const float*)d_in[17];
  const float* cwo       = (const float*)d_in[18];
  float* out = (float*)d_out;

  char* ws = (char*)d_ws;
  size_t off = 0;
  auto alloc = [&](size_t bytes) -> char* {
    char* p = ws + off;
    off += (bytes + 255) & ~(size_t)255;
    return p;
  };
  const long long NTOK = (long long)CB * CS;           // 8192 rows
  const long long NPAT = (long long)CB * CP;           // 2048 rows
  const long long NELT = NTOK * CD;                    // 4.19M
  const long long NFFN = NTOK * CF;

  // bf16 transposed weights
  bf16* wq_t  = (bf16*)alloc((size_t)CD * CD * 2);
  bf16* wk_t  = (bf16*)alloc((size_t)CD * CD * 2);
  bf16* wv_t  = (bf16*)alloc((size_t)CD * CD * 2);
  bf16* wo_t  = (bf16*)alloc((size_t)CD * CD * 2);
  bf16* w1_t  = (bf16*)alloc((size_t)CD * CF * 2);
  bf16* w3_t  = (bf16*)alloc((size_t)CD * CF * 2);
  bf16* w2_t  = (bf16*)alloc((size_t)CD * CF * 2);
  bf16* cwq_t = (bf16*)alloc((size_t)CD * CD * 2);
  bf16* cwk_t = (bf16*)alloc((size_t)CD * CD * 2);
  bf16* cwv_t = (bf16*)alloc((size_t)CD * CD * 2);
  bf16* cwo_t = (bf16*)alloc((size_t)CD * CD * 2);
  // activations
  float* h1   = (float*)alloc(NELT * 4);
  float* h2   = (float*)alloc(NELT * 4);
  float* h3   = (float*)alloc(NELT * 4);
  bf16*  xb   = (bf16*)alloc(NELT * 2);
  float* scrA = (float*)alloc(NFFN * 4);   // q/k/v f32, later g1, later cq/ck/cv
  float* scrB = (float*)alloc(NFFN * 4);   // o f32, later g3, later co
  bf16*  q_bf = (bf16*)alloc(NELT * 2);    // later cq_bf
  bf16*  k_bf = (bf16*)alloc(NELT * 2);    // later ck_bf
  bf16*  vT_bf= (bf16*)alloc(NELT * 2);    // later cvT_bf
  bf16*  o_bf = (bf16*)alloc(NELT * 2);    // later co_bf
  bf16*  gp_bf= (bf16*)alloc(NFFN * 2);
  float* pooled = (float*)alloc((size_t)NPAT * CD * 4);
  int*   cum  = (int*)alloc((size_t)CB * CP * 4);

  float* qf32 = scrA;
  float* kf32 = scrA + NELT;
  float* vf32 = scrA + 2 * NELT;
  float* of32 = scrB;
  float* g1   = scrA;
  float* g3   = scrB;
  float* cqf32 = scrA;
  float* ckf32 = scrA + NELT;
  float* cvf32 = scrA + 2 * NELT;
  float* cof32 = scrB;
  bf16*  qn_bf = q_bf;
  bf16*  kv_bf = xb;
  bf16*  cq_bf = q_bf;   // reuse after self-attn
  bf16*  ck_bf = k_bf;
  bf16*  cvT_bf = vT_bf;
  bf16*  co_bf = o_bf;

  auto gemm = [&](const bf16* A, const bf16* Bt, const float* Csrc, float* C,
                  int M, int N, int K) {
    int tiles = (M / (16 * GEMM_MT)) * (N / (16 * GEMM_NT));
    gemm_wmma_bf16_kernel<<<cdiv(tiles, 8), 256, 0, stream>>>(A, Bt, Csrc, C, M, N, K);
  };
  auto trw = [&](const float* W, bf16* Wt, int K, int N) {
    transpose_cast_w_kernel<<<cdiv((long long)K * N, 256), 256, 0, stream>>>(W, Wt, K, N);
  };

  // ---- weight prep
  trw(wq, wq_t, CD, CD);  trw(wk, wk_t, CD, CD);
  trw(wv, wv_t, CD, CD);  trw(wo, wo_t, CD, CD);
  trw(w1, w1_t, CD, CF);  trw(w3, w3_t, CD, CF);  trw(w2, w2_t, CF, CD);
  trw(cwq, cwq_t, CD, CD); trw(cwk, cwk_t, CD, CD);
  trw(cwv, cwv_t, CD, CD); trw(cwo, cwo_t, CD, CD);

  // ---- embeddings
  hash_embed_kernel<<<(int)NTOK, 128, 0, stream>>>(tokens, tok_emb, hash_emb, h1);

  // ---- self attention
  rms_cast_kernel<<<(int)NTOK, 256, 0, stream>>>(h1, attn_nw, xb, CD);
  gemm(xb, wq_t, nullptr, qf32, (int)NTOK, CD, CD);
  gemm(xb, wk_t, nullptr, kf32, (int)NTOK, CD, CD);
  gemm(xb, wv_t, nullptr, vf32, (int)NTOK, CD, CD);
  long long nrope = NELT / 2;
  rope_cast_kernel<<<cdiv(nrope, 256), 256, 0, stream>>>(qf32, q_bf, CS, nrope);
  rope_cast_kernel<<<cdiv(nrope, 256), 256, 0, stream>>>(kf32, k_bf, CS, nrope);
  transpose_v_kernel<<<cdiv(NELT, 256), 256, 0, stream>>>(vf32, vT_bf, CS, NELT);
  attn_wmma_kernel<<<dim3(CS / 16, CNH, CB), 32, 0, stream>>>(
      q_bf, k_bf, vT_bf, of32, CS, CS, 1);
  cast_bf_kernel<<<cdiv(NELT, 256), 256, 0, stream>>>(of32, o_bf, NELT);
  gemm(o_bf, wo_t, h1, h2, (int)NTOK, CD, CD);

  // ---- FFN
  rms_cast_kernel<<<(int)NTOK, 256, 0, stream>>>(h2, ffn_nw, xb, CD);
  gemm(xb, w1_t, nullptr, g1, (int)NTOK, CF, CD);
  gemm(xb, w3_t, nullptr, g3, (int)NTOK, CF, CD);
  silu_mul_kernel<<<cdiv(NFFN, 256), 256, 0, stream>>>(g1, g3, gp_bf, NFFN);
  gemm(gp_bf, w2_t, h2, h3, (int)NTOK, CD, CF);

  // ---- patch pooling (segment max)
  cumsum_kernel<<<CB, CP, 0, stream>>>(patch_len, cum);
  long long npool = NPAT * CD;
  fill_neg_inf_kernel<<<cdiv(npool, 256), 256, 0, stream>>>(pooled, npool);
  scatter_max_kernel<<<(int)NTOK, 128, 0, stream>>>(h3, cum, pooled);
  fix_pooled_kernel<<<cdiv(npool, 256), 256, 0, stream>>>(pooled, npool);

  // ---- cross attention
  rms_cast_kernel<<<(int)NPAT, 256, 0, stream>>>(pooled, cq_nw, qn_bf, CD);
  gemm(qn_bf, cwq_t, nullptr, cqf32, (int)NPAT, CD, CD);
  rms_cast_kernel<<<(int)NTOK, 256, 0, stream>>>(h3, ckv_nw, kv_bf, CD);
  gemm(kv_bf, cwk_t, nullptr, ckf32, (int)NTOK, CD, CD);
  gemm(kv_bf, cwv_t, nullptr, cvf32, (int)NTOK, CD, CD);
  cast_bf_kernel<<<cdiv(npool, 256), 256, 0, stream>>>(cqf32, cq_bf, npool);
  cast_bf_kernel<<<cdiv(NELT, 256), 256, 0, stream>>>(ckf32, ck_bf, NELT);
  transpose_v_kernel<<<cdiv(NELT, 256), 256, 0, stream>>>(cvf32, cvT_bf, CS, NELT);
  attn_wmma_kernel<<<dim3(CP / 16, CNH, CB), 32, 0, stream>>>(
      cq_bf, ck_bf, cvT_bf, cof32, CP, CS, 0);
  cast_bf_kernel<<<cdiv(npool, 256), 256, 0, stream>>>(cof32, co_bf, npool);

  // ---- final: out = pooled + co @ cwo
  gemm(co_bf, cwo_t, pooled, out, (int)NPAT, CD, CD);
}
